// WindowAttention3d_3435973836949
// MI455X (gfx1250) — matrix operand order repaired and checked by reading, not compiled, not run on hardware
//
#include <hip/hip_runtime.h>
#include <hip/hip_bf16.h>
#include <math.h>
#include <stdint.h>

typedef __attribute__((ext_vector_type(16))) _Float16 v16h;
typedef __attribute__((ext_vector_type(8)))  _Float16 v8h;
typedef __attribute__((ext_vector_type(4)))  _Float16 v4h;
typedef __attribute__((ext_vector_type(8)))  float    v8f;
typedef __attribute__((ext_vector_type(4)))  float    v4f;

#define H_    8
#define HD_   32
#define E_    256
#define L_    343
#define NW_   128
#define ROWS_ (NW_ * L_)   /* 43904 */
#define LP_   352          /* L padded to 16 */
#define LL_   (L_ * L_)    /* 117649 */
#define LLP_  117664       /* LL_ padded to 16-pair tiles */

#define LOG2E_   1.44269504088896340736f
#define MASKVAL_ (-100000.0f * LOG2E_)   /* masked logit, base-2 units */
#define NEGHUGE_ (-3.0e38f)

union AFrag { v16h v; v8h h[2]; };

__device__ __forceinline__ void wave_lds_fence() {
  // wave-local LDS producer->consumer ordering (cross-lane through LDS stage)
  asm volatile("s_wait_dscnt 0" ::: "memory");
}

__device__ __forceinline__ void async_wait0() {
  asm volatile("s_wait_asynccnt 0" ::: "memory");
}

__device__ __forceinline__ void async_copy16_to_lds(uint32_t lds_off,
                                                    const void* gptr) {
  // GLOBAL_LOAD_ASYNC_TO_LDS_B128: per-lane 16B DMA, tracked by ASYNCcnt
  asm volatile("global_load_async_to_lds_b128 %0, %1, off"
               :: "v"(lds_off), "v"((uint64_t)(uintptr_t)gptr)
               : "memory");
}

__device__ __forceinline__ v8f wmma_f16(v16h a, v16h b, v8f c) {
  // D = A(16x32 f16) x B(32x16 f16) + C(16x16 f32)
  return __builtin_amdgcn_wmma_f32_16x16x32_f16(
      false, a, false, b, (short)0, c, false, false);
}

// ---------------------------------------------------------------------------
// Generic E=256 GEMM: (rows x 256) @ (256 x 256) with fused epilogues.
// MODE 0: +bias, per-head L2 normalize, * extra (LOG2E/max(scale,.01) for Q,
//         1.0 for K via scalep==null), f16 headed out
// MODE 2: plain f16 headed out (V, no bias)
// MODE 3: +bias, f32 row-major out (final projection)
// Headed layout: [w*H + h][L][HD] f16.
// ---------------------------------------------------------------------------
template<int MODE, bool A16>
__global__ __launch_bounds__(256)
void gemm_e256(const void* __restrict__ Ap, const float* __restrict__ W,
               const float* __restrict__ bias, void* __restrict__ Out,
               const float* __restrict__ scalep)
{
  __shared__ alignas(32) _Float16 As[128 * 32];   // A tile, row-major
  __shared__ alignas(32) _Float16 Bs[256 * 32];   // B tile, transposed [n][k]
  const int tid  = threadIdx.x;
  const int lane = tid & 31;
  const int wv   = tid >> 5;
  const int bm   = blockIdx.x * 128;

  const float*    A32  = (const float*)Ap;
  const _Float16* A16p = (const _Float16*)Ap;

  v8f acc[16];
  v8f zero = {};
  for (int i = 0; i < 16; ++i) acc[i] = zero;

  const int M    = lane & 15;
  const int koff = (lane >> 4) * 8;    // A-frag half offset
  const int klo  = (lane >> 4) * 16;   // B-frag half offset

  for (int kk = 0; kk < 256; kk += 32) {
    // ---- stage A (128x32) as f16, vectorized ----
    if (A16) {
      for (int t = tid; t < 128 * 32 / 8; t += 256) {   // 2 iters
        int r = t >> 2, c8 = (t & 3) * 8;
        *(v8h*)&As[r * 32 + c8] =
            *(const v8h*)&A16p[(size_t)(bm + r) * 256 + kk + c8];
      }
    } else {
      for (int t = tid; t < 128 * 32 / 4; t += 256) {   // 4 iters
        int r = t >> 3, c4 = (t & 7) * 4;
        v4f x = *(const v4f*)&A32[(size_t)(bm + r) * 256 + kk + c4];
        v4h y;
        for (int e = 0; e < 4; ++e) y[e] = (_Float16)x[e];
        *(v4h*)&As[r * 32 + c4] = y;
      }
    }
    // ---- stage B (32x256) transposed -> Bs[n*32+k], coalesced reads ----
    for (int t = tid; t < 32 * 256 / 4; t += 256) {     // 8 iters
      int k = t >> 6, n4 = (t & 63) * 4;
      v4f x = *(const v4f*)&W[(size_t)(kk + k) * 256 + n4];
      for (int e = 0; e < 4; ++e) Bs[(n4 + e) * 32 + k] = (_Float16)x[e];
    }
    __syncthreads();

    AFrag a;
    a.h[0] = *(const v8h*)&As[(wv * 16 + M) * 32 + 2 * koff];
    a.h[1] = *(const v8h*)&As[(wv * 16 + M) * 32 + 2 * koff + 8];
    for (int nt = 0; nt < 16; ++nt) {
      int n = nt * 16 + (lane & 15);
      v16h b = *(const v16h*)&Bs[n * 32 + klo];
      acc[nt] = wmma_f16(a.v, b, acc[nt]);
    }
    __syncthreads();
  }

  const int rowHi = (lane >> 4) * 8;   // C layout: row = v + rowHi

  if (MODE == 3) {
    float* out = (float*)Out;
    for (int nt = 0; nt < 16; ++nt) {
      int gc = nt * 16 + (lane & 15);
      float bb = bias ? bias[gc] : 0.0f;
      for (int v = 0; v < 8; ++v) {
        int gr = bm + wv * 16 + v + rowHi;
        out[(size_t)gr * 256 + gc] = acc[nt][v] + bb;
      }
    }
  } else {
    if (bias) {
      for (int nt = 0; nt < 16; ++nt) {
        float bb = bias[nt * 16 + (lane & 15)];
        for (int v = 0; v < 8; ++v) acc[nt][v] += bb;
      }
    }
    // Q: fold 1/max(scale,0.01) and LOG2E (softmax in base-2) into q-hat.
    float extra = 1.0f;
    if (MODE == 0 && scalep) extra = LOG2E_ / fmaxf(scalep[0], 0.01f);
    _Float16* outh = (_Float16*)Out;
    for (int h = 0; h < H_; ++h) {
      v8f inv;
      if (MODE == 2) {
        for (int v = 0; v < 8; ++v) inv[v] = 1.0f;
      } else {
        v8f ss;
        for (int v = 0; v < 8; ++v)
          ss[v] = acc[2*h][v] * acc[2*h][v] + acc[2*h+1][v] * acc[2*h+1][v];
        for (int d = 1; d < 16; d <<= 1)
          for (int v = 0; v < 8; ++v)
            ss[v] += __shfl_xor(ss[v], d, 32);
        for (int v = 0; v < 8; ++v)
          inv[v] = extra *
                   __builtin_amdgcn_rcpf(fmaxf(sqrtf(ss[v]), 1e-12f));
      }
      for (int s = 0; s < 2; ++s) {
        int nt = 2 * h + s;
        int cin = s * 16 + (lane & 15);
        for (int v = 0; v < 8; ++v) {
          int gr = bm + wv * 16 + v + rowHi;
          int w = gr / L_, l = gr % L_;
          outh[(((size_t)(w * H_ + h)) * L_ + l) * HD_ + cin] =
              (_Float16)(acc[nt][v] * inv[v]);
        }
      }
    }
  }
}

// ---------------------------------------------------------------------------
// Continuous position bias MLP: log-spaced coords -> 512 relu -> 8, 16*sigmoid.
// One 16-pair tile per wave; layer-2 K=512 via 16 WMMAs against an LDS-resident
// zero-padded transposed W2. Out: [H][LLP_] f32, pre-scaled by LOG2E.
// Stores are guard-free: output is padded to LLP_ per head.
// ---------------------------------------------------------------------------
__global__ __launch_bounds__(256)
void cpb_kernel(const float* __restrict__ idx, const float* __restrict__ W1,
                const float* __restrict__ b1, const float* __restrict__ W2,
                const float* __restrict__ b2, float* __restrict__ biasOut)
{
  __shared__ alignas(16) float    xs[8][16][4];
  __shared__ alignas(32) _Float16 stage[8][16 * 32];
  __shared__ alignas(32) _Float16 W2t[16 * 512];   // [n][k], rows 8..15 zero
  const int tid = threadIdx.x, lane = tid & 31, wv = tid >> 5;
  const int pbase = (blockIdx.x * 8 + wv) * 16;

  // cooperative W2^T stage (all waves participate; barrier-safe)
  for (int t = tid; t < 512; t += 256) {          // 2 iters: k rows
    v4f lo = *(const v4f*)&W2[(size_t)t * 8];
    v4f hi = *(const v4f*)&W2[(size_t)t * 8 + 4];
    for (int e = 0; e < 4; ++e) {
      W2t[e * 512 + t]       = (_Float16)lo[e];
      W2t[(e + 4) * 512 + t] = (_Float16)hi[e];
    }
  }
  {
    v8h z8 = {};
    for (int t = tid; t < 8 * 512 / 8; t += 256)  // pad rows 8..15 = 0
      *(v8h*)&W2t[8 * 512 + t * 8] = z8;
  }
  __syncthreads();

  if (pbase >= LL_) return;   // after the only barrier

  if (lane < 16) {
    int p = pbase + lane; if (p >= LL_) p = LL_ - 1;
    for (int k = 0; k < 3; ++k) {
      float x = idx[(size_t)p * 3 + k];
      // sign(x) * log2(1+|x|) / log2(8)
      xs[wv][lane][k] = copysignf(log2f(1.0f + fabsf(x)) * (1.0f / 3.0f), x);
    }
  }
  wave_lds_fence();

  const int M    = lane & 15;
  const int koff = (lane >> 4) * 8;
  const int N    = lane & 15;
  const int kup  = (lane >> 4) * 16;

  v8f acc = {};
  for (int kb = 0; kb < 512; kb += 32) {
    // layer-1 chunk: 16 rows x 32 hidden, relu, into per-wave LDS stage
    for (int t = lane; t < 16 * 32; t += 32) {
      int r = t >> 5, c32 = t & 31;
      int c = kb + c32;
      float hv = xs[wv][r][0] * W1[c] + xs[wv][r][1] * W1[512 + c] +
                 xs[wv][r][2] * W1[1024 + c] + b1[c];
      stage[wv][t] = (_Float16)fmaxf(hv, 0.0f);
    }
    wave_lds_fence();
    AFrag a;
    a.h[0] = *(const v8h*)&stage[wv][M * 32 + 2 * koff];
    a.h[1] = *(const v8h*)&stage[wv][M * 32 + 2 * koff + 8];
    v16h b = *(const v16h*)&W2t[N * 512 + kb + kup];
    acc = wmma_f16(a.v, b, acc);
    wave_lds_fence();
  }

  if (N < 8) {
    float bb = b2[N];
    for (int v = 0; v < 8; ++v) {
      int p = pbase + v + ((lane >> 4) << 3);   // < LLP_ always
      // 16*LOG2E*sigmoid via fast reciprocal (single v_rcp_f32)
      biasOut[(size_t)N * LLP_ + p] =
          (16.0f * LOG2E_) *
          __builtin_amdgcn_rcpf(1.0f + __expf(-(acc[v] + bb)));
    }
  }
}

// ---------------------------------------------------------------------------
// Flash attention per (window, head): S2 = q̂ k̂ᵀ + bias2 (both pre-scaled by
// LOG2E), mask -> MASKVAL_, base-2 online softmax, O = P V.
// K staged to LDS via async DMA (ASYNCcnt); V transposed manually.
// Bias/mask application is branchless (clamped loads + v_cndmask selects).
// ---------------------------------------------------------------------------
__global__ __launch_bounds__(256)
void attn_kernel(const _Float16* __restrict__ qh, const _Float16* __restrict__ kh,
                 const _Float16* __restrict__ vh, const float* __restrict__ biasp,
                 const unsigned char* __restrict__ maskp,
                 _Float16* __restrict__ outh)
{
  __shared__ alignas(32) _Float16 Ks[LP_ * 32];      // [j][c]
  __shared__ alignas(32) _Float16 Vt[32 * LP_];      // [c][j]
  __shared__ alignas(32) _Float16 stage[8][16 * 32]; // per-wave P tile
  const int tid = threadIdx.x, lane = tid & 31, wv = tid >> 5;
  const int wh = blockIdx.x;             // w*H + h
  const int w = wh >> 3, h = wh & 7;
  const size_t base = (size_t)wh * L_ * HD_;

  // --- K: async DMA global->LDS (16B per lane per op), pad rows zeroed ---
  {
    v8h z8 = {};
    for (int t = tid; t < (LP_ - L_) * 32 / 8; t += 256)   // 36 chunks
      *(v8h*)&Ks[L_ * 32 + t * 8] = z8;
    for (int t = tid; t < L_ * 4; t += 256) {              // 1372 16B chunks
      uint32_t dst = (uint32_t)(uintptr_t)&Ks[t * 8];
      async_copy16_to_lds(dst, kh + base + (size_t)t * 8);
    }
  }
  // --- V: vectorized load + transpose store, pad cols zeroed ---
  for (int t = tid; t < L_ * 4; t += 256) {
    int j = t >> 2, c8 = (t & 3) * 8;
    v8h x = *(const v8h*)&vh[base + (size_t)j * 32 + c8];
    for (int e = 0; e < 8; ++e) Vt[(c8 + e) * LP_ + j] = x[e];
  }
  for (int t = tid; t < 32 * (LP_ - L_); t += 256) {       // 288 elems
    int c = t / (LP_ - L_), j = L_ + (t % (LP_ - L_));
    Vt[c * LP_ + j] = (_Float16)0.0f;
  }
  async_wait0();
  __syncthreads();

  const float* biasH         = biasp + (size_t)h * LLP_;
  const unsigned char* maskW = maskp + (size_t)w * LL_;

  const int M     = lane & 15;
  const int koff  = (lane >> 4) * 8;
  const int kup16 = (lane >> 4) * 16;
  const int rowHi = (lane >> 4) * 8;

  for (int mt = wv; mt < 22; mt += 8) {        // 22 row tiles of 16 (352)
    const int row0 = mt * 16;
    AFrag aq;
    {
      int r = row0 + M; if (r >= L_) r = L_ - 1;
      aq.h[0] = *(const v8h*)&qh[base + (size_t)r * 32 + 2 * koff];
      aq.h[1] = *(const v8h*)&qh[base + (size_t)r * 32 + 2 * koff + 8];
    }
    v8f O0 = {}, O1 = {};
    float mrow[8], lrow[8];
    for (int v = 0; v < 8; ++v) { mrow[v] = NEGHUGE_; lrow[v] = 0.0f; }

    for (int jt = 0; jt < 11; ++jt) {          // 11 chunks of 32 keys
      const int jb = jt * 32;
      v16h bk0, bk1;
      {
        int j0 = jb + (lane & 15);
        bk0 = *(const v16h*)&Ks[j0 * 32 + kup16];
        bk1 = *(const v16h*)&Ks[(j0 + 16) * 32 + kup16];
      }
      v8f zero = {};
      v8f S0 = wmma_f16(aq.v, bk0, zero);
      v8f S1 = wmma_f16(aq.v, bk1, zero);

      // branchless bias + mask (clamped loads, cndmask selects)
      {
        const int j0 = jb + (lane & 15);
        const int j1 = j0 + 16;
        const int jc0 = (j0 < L_) ? j0 : (L_ - 1);
        const int jc1 = (j1 < L_) ? j1 : (L_ - 1);
        for (int v = 0; v < 8; ++v) {
          int i  = row0 + v + rowHi;
          int ic = (i < L_) ? i : (L_ - 1);
          const size_t r0 = (size_t)ic * L_;
          float b0 = biasH[r0 + jc0];
          float b1v = biasH[r0 + jc1];
          bool m0 = maskW[r0 + jc0] != 0;
          bool m1 = maskW[r0 + jc1] != 0;
          float s0 = S0[v] + b0;
          float s1 = S1[v] + b1v;
          s0 = m0 ? MASKVAL_ : s0;
          s1 = m1 ? MASKVAL_ : s1;
          s0 = (j0 < L_) ? s0 : NEGHUGE_;
          s1 = (j1 < L_) ? s1 : NEGHUGE_;
          S0[v] = s0; S1[v] = s1;
        }
      }

      // base-2 online softmax: row max / rescale / row sum (butterflies)
      float f[8];
      for (int v = 0; v < 8; ++v) {
        float t = fmaxf(S0[v], S1[v]);
        for (int d = 1; d < 16; d <<= 1) t = fmaxf(t, __shfl_xor(t, d, 32));
        float mn = fmaxf(mrow[v], t);
        f[v] = __builtin_amdgcn_exp2f(mrow[v] - mn);
        mrow[v] = mn;
      }
      for (int v = 0; v < 8; ++v) {
        float p0 = __builtin_amdgcn_exp2f(S0[v] - mrow[v]);
        float p1 = __builtin_amdgcn_exp2f(S1[v] - mrow[v]);
        S0[v] = p0; S1[v] = p1;
        float rs = p0 + p1;
        for (int d = 1; d < 16; d <<= 1) rs += __shfl_xor(rs, d, 32);
        lrow[v] = lrow[v] * f[v] + rs;
        O0[v] *= f[v]; O1[v] *= f[v];
      }

      // C-layout P -> A-layout via per-wave LDS stage
      for (int v = 0; v < 8; ++v) {
        int r = v + rowHi;
        stage[wv][r * 32 + (lane & 15)]      = (_Float16)S0[v];
        stage[wv][r * 32 + 16 + (lane & 15)] = (_Float16)S1[v];
      }
      wave_lds_fence();
      AFrag ap;
      ap.h[0] = *(const v8h*)&stage[wv][M * 32 + 2 * koff];
      ap.h[1] = *(const v8h*)&stage[wv][M * 32 + 2 * koff + 8];
      v16h bv0, bv1;
      {
        int n = lane & 15;
        bv0 = *(const v16h*)&Vt[n * LP_ + jb + kup16];
        bv1 = *(const v16h*)&Vt[(n + 16) * LP_ + jb + kup16];
      }
      O0 = wmma_f16(ap.v, bv0, O0);
      O1 = wmma_f16(ap.v, bv1, O1);
      wave_lds_fence();
    }

    for (int v = 0; v < 8; ++v) {
      int i = row0 + v + rowHi;
      if (i < L_) {
        float inv = __builtin_amdgcn_rcpf(lrow[v]);
        size_t o = ((size_t)(w * L_ + i)) * 256 + h * 32;
        outh[o + (lane & 15)]      = (_Float16)(O0[v] * inv);
        outh[o + 16 + (lane & 15)] = (_Float16)(O1[v] * inv);
      }
    }
  }
}

// ---------------------------------------------------------------------------
extern "C" void kernel_launch(void* const* d_in, const int* in_sizes, int n_in,
                              void* d_out, int out_size, void* d_ws, size_t ws_size,
                              hipStream_t stream) {
  const float* q     = (const float*)d_in[0];
  const float* k     = (const float*)d_in[1];
  const float* v     = (const float*)d_in[2];
  const float* idx   = (const float*)d_in[3];
  const unsigned char* mask = (const unsigned char*)d_in[4];
  const float* Wq    = (const float*)d_in[5];
  const float* bq    = (const float*)d_in[6];
  const float* Wk    = (const float*)d_in[7];
  const float* bk    = (const float*)d_in[8];
  const float* Wv    = (const float*)d_in[9];
  const float* Ww    = (const float*)d_in[10];
  const float* bw    = (const float*)d_in[11];
  const float* scale = (const float*)d_in[12];
  const float* W1    = (const float*)d_in[13];
  const float* b1    = (const float*)d_in[14];
  const float* W2    = (const float*)d_in[15];
  const float* b2    = (const float*)d_in[16];

  char* ws = (char*)d_ws;
  size_t off = 0;
  auto take = [&](size_t bytes) {
    char* p = ws + off;
    off = (off + bytes + 255) & ~(size_t)255;
    return p;
  };
  _Float16* qh      = (_Float16*)take((size_t)ROWS_ * 256 * 2);
  _Float16* khd     = (_Float16*)take((size_t)ROWS_ * 256 * 2);
  _Float16* vhd     = (_Float16*)take((size_t)ROWS_ * 256 * 2);
  float*    biasws  = (float*)take((size_t)H_ * LLP_ * 4);
  _Float16* attnout = (_Float16*)take((size_t)ROWS_ * 256 * 2);

  dim3 blk(256);
  // QKV projections with fused bias + L2-norm (+LOG2E/scale for Q), f16 out
  gemm_e256<0, false><<<dim3(ROWS_ / 128), blk, 0, stream>>>(q, Wq, bq, qh, scale);
  gemm_e256<0, false><<<dim3(ROWS_ / 128), blk, 0, stream>>>(k, Wk, bk, khd, nullptr);
  gemm_e256<2, false><<<dim3(ROWS_ / 128), blk, 0, stream>>>(v, Wv, nullptr, vhd, nullptr);
  // CPB bias table (computed once, L2-resident for all 128 windows)
  int cpb_tiles  = (LL_ + 15) / 16;
  int cpb_blocks = (cpb_tiles + 7) / 8;
  cpb_kernel<<<dim3(cpb_blocks), blk, 0, stream>>>(idx, W1, b1, W2, b2, biasws);
  // Flash attention per (window, head)
  attn_kernel<<<dim3(NW_ * H_), blk, 0, stream>>>(qh, khd, vhd, biasws, mask, attnout);
  // Output projection -> f32
  gemm_e256<3, true><<<dim3(ROWS_ / 128), blk, 0, stream>>>(attnout, Ww, bw, (float*)d_out, nullptr);
}
